// RuchbahMamba3Block_14963666059312
// MI455X (gfx1250) — compile-verified, matrix operand order checked by reading, not run on hardware
//
#include <hip/hip_runtime.h>
#include <hip/hip_bf16.h>
#include <math.h>

// ---------------------------------------------------------------------------
// Mamba block forward for MI455X (gfx1250), wave32, bf16 WMMA path.
//   B=2, L=4096, DM=1024, DI=2048, DT=64, K=4
//
// Roofline: ~107 GFLOP of GEMM vs ~0.5 GB of HBM traffic -> memory bound.
// Therefore: bf16 storage for all intermediate activations (xz, xi, dt_low, y)
// to halve traffic; f32 kept only for the scan decay factor `e` (error
// compounds over 4096 sequential steps) and the final output.
//
// Pipeline:
//   1) gemm : xz(bf16)   = x(f32) @ in_proj_w^T            [8192, 4096]
//   2) conv : xi(bf16)   = silu(causal_dwconv(xz[:, :DI]))  [8192, 2048]
//   3) gemm : dtl(bf16)  = xi @ x_proj_w^T                  [8192, 64]
//   4) gemm : e(f32)     = decay_epilogue(dtl @ dt_proj_w^T)[8192, 2048]
//   5) scan : y(bf16)    = clip(scan(e, xi) + xi*D) * silu(z)
//   6) gemm : out(f32)   = y @ out_proj_w^T                 [8192, 1024]
// ---------------------------------------------------------------------------

typedef __attribute__((ext_vector_type(16))) __bf16 v16bf;
typedef __attribute__((ext_vector_type(8)))  __bf16 v8bf;
typedef __attribute__((ext_vector_type(8)))  float  v8f;
typedef __attribute__((ext_vector_type(4)))  float  v4f;

#define MB_B   2
#define MB_L   4096
#define MB_DM  1024
#define MB_DI  2048
#define MB_DT  64
#define MB_K   4
#define MB_M   (MB_B * MB_L)   // 8192 rows

// LDS tile pitch in bf16 elements: 32 data + 8 pad (keeps the 8-element
// fragment runs 16B-aligned, staggers rows across banks).
#define LDS_PITCH 40

#define EPI_NONE  0
#define EPI_DECAY 1

// ---- small helpers --------------------------------------------------------
__device__ inline v8bf ld8_bf16(const float* p) {   // load 8 f32, cvt -> 8 bf16
    v4f lo = *(const v4f*)p;
    v4f hi = *(const v4f*)(p + 4);
    v8bf r;
#pragma unroll
    for (int i = 0; i < 4; ++i) { r[i] = (__bf16)lo[i]; r[4 + i] = (__bf16)hi[i]; }
    return r;
}
__device__ inline v8bf ld8_bf16(const __bf16* p) {  // direct 16B copy
    return *(const v8bf*)p;
}
__device__ inline void st_out(float*  p, float v) { *p = v; }
__device__ inline void st_out(__bf16* p, float v) { *p = (__bf16)v; }

// ---------------------------------------------------------------------------
// O[M,N] = epilogue( A[M,K] * W[N,K]^T )   (bf16 WMMA, f32 accumulate)
//   A: f32 or bf16 row-major;  W: f32 row-major [N,K];  O: f32 or bf16.
// Block: 128 threads = 4 waves. Block tile 128(M) x 64(N), K-step 32.
// Wave w owns rows [32w, 32w+32): 2 M-subtiles x 4 N-subtiles = 8 WMMA/step,
// with each B fragment reused across both M-subtiles.
// ---------------------------------------------------------------------------
template <typename TA, typename TO>
__launch_bounds__(128)
__global__ void gemm_wmma(const TA*    __restrict__ Amat,
                          const float* __restrict__ Wmat,
                          TO*          __restrict__ Omat,
                          int M, int N, int K,
                          int mode,
                          const float* __restrict__ bias,
                          const float* __restrict__ alog)
{
    __shared__ __align__(16) __bf16 lA[128 * LDS_PITCH];  // 10.0 KB
    __shared__ __align__(16) __bf16 lB[ 64 * LDS_PITCH];  //  5.0 KB

    const int tid  = threadIdx.x;        // 0..127
    const int wave = tid >> 5;           // 0..3
    const int lane = tid & 31;
    const int half = lane >> 4;          // K-half selector (ISA 7.12.2)
    const int l16  = lane & 15;

    const int tileM = blockIdx.x * 128;
    const int tileN = blockIdx.y * 64;

    // B staging: 2 threads per weight row, 16 cols each.
    const int brow = tid >> 1;           // 0..63
    const int bcol = (tid & 1) << 4;     // 0 | 16

    v8f acc[2][4] = {};

    for (int k0 = 0; k0 < K; k0 += 32) {
        // ---- stage A (one full 32-col row per thread) ---------------------
        const TA*    ga = Amat + (size_t)(tileM + tid)  * K + k0;
        const float* gb = Wmat + (size_t)(tileN + brow) * K + (k0 + bcol);

        if (k0 + 32 < K) {               // prefetch next K-slab
            __builtin_prefetch(ga + 32, 0, 1);
            __builtin_prefetch(gb + 32, 0, 1);
        }

        v8bf a0 = ld8_bf16(ga);
        v8bf a1 = ld8_bf16(ga + 8);
        v8bf a2 = ld8_bf16(ga + 16);
        v8bf a3 = ld8_bf16(ga + 24);
        v8bf b0 = ld8_bf16(gb);
        v8bf b1 = ld8_bf16(gb + 8);

        __bf16* sa = &lA[tid * LDS_PITCH];
        *(v8bf*)(sa +  0) = a0;
        *(v8bf*)(sa +  8) = a1;
        *(v8bf*)(sa + 16) = a2;
        *(v8bf*)(sa + 24) = a3;
        __bf16* sb = &lB[brow * LDS_PITCH + bcol];
        *(v8bf*)(sb)     = b0;
        *(v8bf*)(sb + 8) = b1;
        __syncthreads();

        // ---- fragments + WMMA --------------------------------------------
        // 16-bit A 16x32 fragment: lane m = lane&15,
        //   elems[0..7]  = K[half*8 .. +7], elems[8..15] = K[16+half*8 .. +7]
        union { v16bf v; v8bf h[2]; } af[2];
#pragma unroll
        for (int mt = 0; mt < 2; ++mt) {
            const int mrow = wave * 32 + mt * 16 + l16;
            af[mt].h[0] = *(const v8bf*)&lA[mrow * LDS_PITCH + half * 8];
            af[mt].h[1] = *(const v8bf*)&lA[mrow * LDS_PITCH + 16 + half * 8];
        }
#pragma unroll
        for (int nt = 0; nt < 4; ++nt) {
            const int nrow = nt * 16 + l16;
            union { v16bf v; v8bf h[2]; } bf;
            bf.h[0] = *(const v8bf*)&lB[nrow * LDS_PITCH + half * 8];
            bf.h[1] = *(const v8bf*)&lB[nrow * LDS_PITCH + 16 + half * 8];
#pragma unroll
            for (int mt = 0; mt < 2; ++mt) {
                acc[mt][nt] = __builtin_amdgcn_wmma_f32_16x16x32_bf16(
                    false, af[mt].v, false, bf.v, (short)0, acc[mt][nt],
                    false, false);
            }
        }
        __syncthreads();
    }

    // ---- epilogue + store -------------------------------------------------
    // D layout: VGPR r, lanes 0-15 -> (M=r, N=lane); lanes 16-31 -> (M=8+r,..)
#pragma unroll
    for (int nt = 0; nt < 4; ++nt) {
        const int n = tileN + nt * 16 + l16;
        float bn = 0.f, an = 0.f;
        if (mode == EPI_DECAY) {
            bn = bias[n];
            an = fminf(fmaxf(-__expf(alog[n]), -10.f), -1e-6f);
        }
#pragma unroll
        for (int mt = 0; mt < 2; ++mt) {
#pragma unroll
            for (int r = 0; r < 8; ++r) {
                const int m = tileM + wave * 32 + mt * 16 + r + 8 * half;
                float v = acc[mt][nt][r];
                if (mode == EPI_DECAY) {
                    float dlt = v + bn;
                    float sp  = (dlt > 20.f) ? dlt : log1pf(__expf(dlt));
                    sp = fminf(fmaxf(sp, 1e-6f), 10.f);
                    float e = __expf(sp * an);
                    v = fminf(fmaxf(e, 1e-6f), 1.f);
                }
                st_out(&Omat[(size_t)m * N + n], v);
            }
        }
    }
}

// ---------------------------------------------------------------------------
// Depthwise causal conv1d (K=4) + bias + SiLU.  xz is bf16 [M, 2*DI]; the xi
// half is columns [0, DI). One thread per (b, d, 256-step l-chunk), rolling
// window; loads coalesced across d.
// ---------------------------------------------------------------------------
__launch_bounds__(256)
__global__ void conv_silu(const __bf16* __restrict__ xz,
                          const float*  __restrict__ cw,   // [DI, 1, 4]
                          const float*  __restrict__ cb,   // [DI]
                          __bf16*       __restrict__ xi)   // [M, DI] bf16
{
    const int d  = blockIdx.x * 256 + threadIdx.x;  // 0..DI-1
    const int l0 = blockIdx.y * 256;
    const int b  = blockIdx.z;

    const float w0 = cw[d * MB_K + 0];
    const float w1 = cw[d * MB_K + 1];
    const float w2 = cw[d * MB_K + 2];
    const float w3 = cw[d * MB_K + 3];
    const float bb = cb[d];

    const size_t rowbase = (size_t)b * MB_L;
    auto ld = [&](int l) -> float {
        return (l < 0) ? 0.f
                       : (float)xz[(rowbase + l) * (2 * (size_t)MB_DI) + d];
    };

    float xm3 = ld(l0 - 3), xm2 = ld(l0 - 2), xm1 = ld(l0 - 1);
#pragma unroll 4
    for (int l = l0; l < l0 + 256; ++l) {
        const float x0 = ld(l);
        const float v  = w0 * xm3 + w1 * xm2 + w2 * xm1 + w3 * x0 + bb;
        const float s  = v / (1.f + __expf(-v));            // SiLU
        xi[(rowbase + l) * MB_DI + d] = (__bf16)s;
        xm3 = xm2; xm2 = xm1; xm1 = x0;
    }
}

// ---------------------------------------------------------------------------
// Selective scan + skip + gate. One thread per (b, d): serial over L, per-wave
// coalesced loads (consecutive d -> consecutive addresses at each l).
//   state = clip(state*e + u, ±1e4);  y = clip(state + u*D, ±1e4) * silu(z)
// State kept in f32; e kept f32 since its error compounds over L steps.
// ---------------------------------------------------------------------------
__launch_bounds__(256)
__global__ void scan_gate(const float*  __restrict__ ebuf,  // [M, DI] f32
                          const __bf16* __restrict__ xi,    // [M, DI] bf16
                          const __bf16* __restrict__ xz,    // [M, 2*DI] (z half)
                          const float*  __restrict__ Dv,    // [DI]
                          __bf16*       __restrict__ y)     // [M, DI] bf16
{
    const int d = blockIdx.x * 256 + threadIdx.x;  // 0..DI-1
    const int b = blockIdx.y;
    const float Dd = Dv[d];
    const size_t rowbase = (size_t)b * MB_L;

    float state = 0.f;
#pragma unroll 4
    for (int l = 0; l < MB_L; ++l) {
        const size_t m = rowbase + l;
        const float et = ebuf[m * MB_DI + d];
        const float ut = (float)xi[m * MB_DI + d];
        const float zz = (float)xz[m * (2 * (size_t)MB_DI) + MB_DI + d];
        state = fminf(fmaxf(state * et + ut, -1e4f), 1e4f);
        float yy = fminf(fmaxf(state + ut * Dd, -1e4f), 1e4f);
        const float g = zz / (1.f + __expf(-zz));            // silu(z)
        y[m * MB_DI + d] = (__bf16)(yy * g);
    }
}

// ---------------------------------------------------------------------------
extern "C" void kernel_launch(void* const* d_in, const int* in_sizes, int n_in,
                              void* d_out, int out_size, void* d_ws, size_t ws_size,
                              hipStream_t stream) {
    const float* x          = (const float*)d_in[0];  // [B,L,DM]
    const float* in_proj_w  = (const float*)d_in[1];  // [2*DI, DM]
    const float* conv_w     = (const float*)d_in[2];  // [DI,1,K]
    const float* conv_b     = (const float*)d_in[3];  // [DI]
    const float* x_proj_w   = (const float*)d_in[4];  // [DT, DI]
    const float* dt_proj_w  = (const float*)d_in[5];  // [DI, DT]
    const float* dt_proj_b  = (const float*)d_in[6];  // [DI]
    const float* A_log      = (const float*)d_in[7];  // [DI]
    const float* Dvec       = (const float*)d_in[8];  // [DI]
    const float* out_proj_w = (const float*)d_in[9];  // [DM, DI]
    float* out = (float*)d_out;                       // [B,L,DM] f32

    // Workspace (bytes): xz 64M | xi 32M | e 64M | dtl 1M | y 32M  (~193 MB)
    __bf16* xz   = (__bf16*)d_ws;                          // M * 2*DI bf16
    __bf16* xi   = xz  + (size_t)MB_M * 2 * MB_DI;         // M * DI   bf16
    float*  ebuf = (float*)(xi + (size_t)MB_M * MB_DI);    // M * DI   f32
    __bf16* dtl  = (__bf16*)(ebuf + (size_t)MB_M * MB_DI); // M * DT   bf16
    __bf16* ybuf = dtl + (size_t)MB_M * MB_DT;             // M * DI   bf16

    const dim3 gblk(128);

    // 1) in_proj: xz = x @ in_proj_w^T   (f32 in, bf16 out)
    gemm_wmma<float, __bf16><<<dim3(MB_M / 128, (2 * MB_DI) / 64), gblk, 0, stream>>>(
        x, in_proj_w, xz, MB_M, 2 * MB_DI, MB_DM, EPI_NONE, nullptr, nullptr);

    // 2) depthwise causal conv + SiLU
    conv_silu<<<dim3(MB_DI / 256, MB_L / 256, MB_B), 256, 0, stream>>>(
        xz, conv_w, conv_b, xi);

    // 3) x_proj: dtl = xi @ x_proj_w^T   (bf16 in, bf16 out)
    gemm_wmma<__bf16, __bf16><<<dim3(MB_M / 128, MB_DT / 64), gblk, 0, stream>>>(
        xi, x_proj_w, dtl, MB_M, MB_DT, MB_DI, EPI_NONE, nullptr, nullptr);

    // 4) dt_proj + fused softplus/exp decay: e (f32)
    gemm_wmma<__bf16, float><<<dim3(MB_M / 128, MB_DI / 64), gblk, 0, stream>>>(
        dtl, dt_proj_w, ebuf, MB_M, MB_DI, MB_DT, EPI_DECAY, dt_proj_b, A_log);

    // 5) selective scan + skip + gate: y (bf16)
    scan_gate<<<dim3(MB_DI / 256, MB_B), 256, 0, stream>>>(
        ebuf, xi, xz, Dvec, ybuf);

    // 6) out_proj: out = y @ out_proj_w^T   (bf16 in, f32 out)
    gemm_wmma<__bf16, float><<<dim3(MB_M / 128, MB_DM / 64), gblk, 0, stream>>>(
        ybuf, out_proj_w, out, MB_M, MB_DM, MB_DI, EPI_NONE, nullptr, nullptr);
}
